// SpGraphAttentionLayer_83932250899045
// MI455X (gfx1250) — matrix operand order repaired and checked by reading, compiled, not run on hardware
//
#include <hip/hip_runtime.h>
#include <hip/hip_bf16.h>

typedef __attribute__((ext_vector_type(16))) __bf16 v16bf;
typedef __attribute__((ext_vector_type(4)))  __bf16 v4bf;
typedef __attribute__((ext_vector_type(8)))  float  v8f;

#define N_NODES 50000
#define IN_F    128
#define OUT_F   200
#define N_E1    400000
#define N_E2    100000
#define E_TOT   (N_E1 + N_E2 + N_NODES)     // 550000 = 80 * 6875 exactly
#define N_TILES 13                          // ceil(200/16)
#define K_STEPS 8                           // 256/32
#define OUT_PAD (N_TILES * 16)              // 208
#define ALPHA   0.2f

#define EDGES_PB 80                         // edges per block (5 M-tiles)
#define MT_PB    5
#define NWAVES   5
#define THREADS  (NWAVES * 32)              // 160

// workspace layout (bytes)
#define A_BF16_ELEMS (N_TILES * K_STEPS * 32 * 16)        // 53248 bf16
#define A_BF16_BYTES (A_BF16_ELEMS * 2)                   // 106496 (256-aligned)
#define NUM_OFF      A_BF16_BYTES
#define NUM_ELEMS    ((size_t)N_NODES * OUT_F)            // 10,000,000 floats
#define ROW_OFF      (NUM_OFF + NUM_ELEMS * 4)            // rowsum right after num

// ---------------------------------------------------------------------------
// Kernel 0: repack a[200][256] fp32 -> bf16 in WMMA B-fragment order.
//   [t][s][lane][e] : B[k][n] = a[n][k]
//   n = t*16 + (lane & 15), k = s*32 + (e & 7) + 16*(e >> 3) + 8*(lane >> 4)   (n >= 200 -> 0)
// ---------------------------------------------------------------------------
__global__ __launch_bounds__(256)
void convert_a_kernel(const float* __restrict__ a, __bf16* __restrict__ aB) {
    int idx = blockIdx.x * 256 + threadIdx.x;
    if (idx >= A_BF16_ELEMS) return;
    int e    = idx & 15;
    int lane = (idx >> 4) & 31;
    int s    = (idx >> 9) & 7;
    int t    = idx >> 12;
    int n = t * 16 + (lane & 15);
    int k = s * 32 + (e & 7) + ((e >> 3) << 4) + 8 * (lane >> 4);
    float v = (n < OUT_F) ? a[n * (2 * IN_F) + k] : 0.0f;
    aB[idx] = (__bf16)v;
}

// ---------------------------------------------------------------------------
// Kernel 1: zero numerator + rowsum accumulators (contiguous region)
// ---------------------------------------------------------------------------
__global__ __launch_bounds__(256)
void zero_ws_kernel(float* __restrict__ p, size_t n) {
    size_t stride = (size_t)gridDim.x * blockDim.x;
    for (size_t i = blockIdx.x * (size_t)blockDim.x + threadIdx.x; i < n; i += stride)
        p[i] = 0.0f;
}

// ---------------------------------------------------------------------------
// Kernel 2: fused edge GEMM (WMMA bf16) + attention weights + scatter-add.
// 160 threads (5 waves) handle 80 edges (5 M-tiles).
// Each wave register-caches the B fragments of one n-tile (64 VGPRs) and
// sweeps all 5 M-tiles with them, amortizing L2 B-traffic 5x.
// ---------------------------------------------------------------------------
__global__ __launch_bounds__(THREADS)
void gat_edge_kernel(const float* __restrict__ input,
                     const int*   __restrict__ edge,        // [2][400000]
                     const float* __restrict__ edge_embed,  // [400000][128]
                     const int*   __restrict__ nhop,        // [2][100000]
                     const float* __restrict__ embed_nhop,  // [100000][128]
                     const float* __restrict__ a2,          // [200]
                     const __bf16* __restrict__ aB,         // pre-swizzled B
                     float* __restrict__ num,               // [N][200] accum
                     float* __restrict__ rowsum)            // [N] accum
{
    __shared__ alignas(32) __bf16 ldsA[MT_PB][K_STEPS][32][16]; // 40 KB
    __shared__ float em[EDGES_PB][OUT_PAD];                     // 66.5 KB
    __shared__ int          srcn[EDGES_PB];
    __shared__ int          tgtn[EDGES_PB];
    __shared__ const float* embp[EDGES_PB];
    __shared__ float        lw[EDGES_PB];

    const int tid = threadIdx.x;
    const int e0  = blockIdx.x * EDGES_PB;

    // --- edge metadata (E_TOT % 80 == 0 -> no tail) ------------------------
    if (tid < EDGES_PB) {
        int e = e0 + tid;
        int s, t; const float* p;
        if (e < N_E1)              { s = edge[e];            t = edge[N_E1 + e];
                                     p = edge_embed + (size_t)e * IN_F; }
        else if (e < N_E1 + N_E2)  { int i = e - N_E1;
                                     s = nhop[i];            t = nhop[N_E2 + i];
                                     p = embed_nhop + (size_t)i * IN_F; }
        else                       { int v = e - N_E1 - N_E2; s = v; t = v; p = nullptr; }
        srcn[tid] = s; tgtn[tid] = t; embp[tid] = p;
    }
    __syncthreads();

    // --- stage edge_h (80 x 256) into LDS in WMMA A-fragment order --------
    // A 16x32 bf16 layout: lane = M + 16*h ; K=(e<8?e:e+8)+8*h.
    // Any 4-aligned feature quad -> 4 contiguous bf16 in one lane fragment.
    for (int q = tid; q < EDGES_PB * 64; q += THREADS) {   // 64 quads per edge
        int m  = q >> 6;
        int i0 = (q & 63) * 4;
        float4 v;
        if (i0 < IN_F) {
            v = *(const float4*)&input[(size_t)tgtn[m] * IN_F + i0];
        } else {
            const float* p = embp[m];
            if (p) v = *(const float4*)&p[i0 - IN_F];
            else   v = make_float4(0.f, 0.f, 0.f, 0.f);
        }
        int mt   = m >> 4;
        int mrow = m & 15;
        int s    = i0 >> 5;
        int ko   = i0 & 31;
        int h    = (ko >> 3) & 1;
        int ebase= (ko & 7) + ((ko >> 4) << 3);            // in {0,4,8,12}
        v4bf pk  = { (__bf16)v.x, (__bf16)v.y, (__bf16)v.z, (__bf16)v.w };
        *(v4bf*)&ldsA[mt][s][mrow + 16 * h][ebase] = pk;
    }
    __syncthreads();

    // --- GEMM: wave owns n-tiles t = wave, wave+5, wave+10 -----------------
    const int wave = tid >> 5;
    const int lane = tid & 31;
    const v16bf* B = (const v16bf*)aB;

    for (int t = wave; t < N_TILES; t += NWAVES) {
        v16bf bfr[K_STEPS];
        #pragma unroll
        for (int s = 0; s < K_STEPS; ++s)
            bfr[s] = B[(t * K_STEPS + s) * 32 + lane];

        #pragma unroll
        for (int mt = 0; mt < MT_PB; ++mt) {
            v8f acc = {};
            #pragma unroll
            for (int s = 0; s < K_STEPS; ++s) {
                v16bf af = *(const v16bf*)&ldsA[mt][s][lane][0];
                acc = __builtin_amdgcn_wmma_f32_16x16x32_bf16(
                          false, af, false, bfr[s], (short)0, acc, false, false);
            }
            // D layout: lane -> N = t*16 + (lane&15); VGPR r -> M = r + 8*(lane>>4)
            int n     = t * 16 + (lane & 15);
            int rbase = mt * 16 + (lane >> 4) * 8;
            #pragma unroll
            for (int r = 0; r < 8; ++r)
                em[rbase + r][n] = acc[r];
        }
    }
    __syncthreads();

    // --- attention weight per edge: w = exp(-leakyrelu(em . a2)) ----------
    if (tid < EDGES_PB) {
        float sc = 0.0f;
        for (int o = 0; o < OUT_F; ++o)
            sc = fmaf(em[tid][o], a2[o], sc);
        float p = (sc >= 0.0f) ? sc : ALPHA * sc;
        float w = __expf(-p);
        lw[tid] = w;
        atomicAdd(&rowsum[srcn[tid]], w);
    }
    __syncthreads();

    // --- numerator scatter: num[src] += w * em -----------------------------
    for (int idx = tid; idx < EDGES_PB * OUT_F; idx += THREADS) {
        int m = idx / OUT_F;
        int o = idx - m * OUT_F;
        atomicAdd(&num[(size_t)srcn[m] * OUT_F + o], lw[m] * em[m][o]);
    }
}

// ---------------------------------------------------------------------------
// Kernel 3: h' = num / rowsum  (0 -> 1e-12), then ELU
// ---------------------------------------------------------------------------
__global__ __launch_bounds__(256)
void finalize_kernel(const float* __restrict__ num,
                     const float* __restrict__ rowsum,
                     float* __restrict__ out)
{
    size_t i = (size_t)blockIdx.x * 256 + threadIdx.x;
    if (i >= NUM_ELEMS) return;
    int node = (int)(i / OUT_F);
    float r = rowsum[node];
    if (r == 0.0f) r = 1e-12f;
    float x = num[i] / r;
    out[i] = (x > 0.0f) ? x : (__expf(x) - 1.0f);
}

// ---------------------------------------------------------------------------
extern "C" void kernel_launch(void* const* d_in, const int* in_sizes, int n_in,
                              void* d_out, int out_size, void* d_ws, size_t ws_size,
                              hipStream_t stream) {
    const float* input      = (const float*)d_in[0];
    const int*   edge       = (const int*)  d_in[1];
    const float* edge_embed = (const float*)d_in[2];
    const int*   nhop       = (const int*)  d_in[3];
    const float* embed_nhop = (const float*)d_in[4];
    const float* a          = (const float*)d_in[5];
    const float* a2         = (const float*)d_in[6];
    float*       out        = (float*)d_out;

    char*   ws     = (char*)d_ws;
    __bf16* aB     = (__bf16*)ws;
    float*  num    = (float*)(ws + NUM_OFF);
    float*  rowsum = (float*)(ws + ROW_OFF);

    convert_a_kernel<<<(A_BF16_ELEMS + 255) / 256, 256, 0, stream>>>(a, aB);
    zero_ws_kernel<<<2048, 256, 0, stream>>>(num, NUM_ELEMS + N_NODES);
    gat_edge_kernel<<<E_TOT / EDGES_PB, THREADS, 0, stream>>>(
        input, edge, edge_embed, nhop, embed_nhop, a2, aB, num, rowsum);
    finalize_kernel<<<(int)((NUM_ELEMS + 255) / 256), 256, 0, stream>>>(num, rowsum, out);
}
